// LinearVQ_26938034881021
// MI455X (gfx1250) — compile-verified, matrix operand order checked by reading, not run on hardware
//
#include <hip/hip_runtime.h>
#include <hip/hip_bf16.h>

// Problem sizes (fixed by the reference)
#define D_DIM 1024
#define R_DIM 256
#define K_DIM 8192
#define N_DIM 32768
#define BETA_F 0.25f

typedef float        v2f   __attribute__((ext_vector_type(2)));
typedef float        v8f   __attribute__((ext_vector_type(8)));
typedef __bf16       v8bf  __attribute__((ext_vector_type(8)));
typedef __bf16       v16bf __attribute__((ext_vector_type(16)));
typedef unsigned int v4u   __attribute__((ext_vector_type(4)));
typedef int          v4i   __attribute__((ext_vector_type(4)));
typedef int          v8i   __attribute__((ext_vector_type(8)));

// CDNA5 native f32 matrix op: D(16x16,f32) = A(16x4,f32) * B(4x16,f32) + C
__device__ __forceinline__ v8f wmma_f32x4(v2f a, v2f b, v8f c) {
    return __builtin_amdgcn_wmma_f32_16x16x4_f32(false, a, false, b, (short)0, c, false, false);
}

// CDNA5 bf16 matrix op: D(16x16,f32) = A(16x32,bf16) * B(32x16,bf16) + C
__device__ __forceinline__ v8f wmma_bf16x32(v16bf a, v16bf b, v8f c) {
    return __builtin_amdgcn_wmma_f32_16x16x32_bf16(false, a, false, b, (short)0, c, false, false);
}

// 16-bit fragment per ISA layout: lane half sel picks K {0..7,16..23} vs {8..15,24..31}
// base already includes row offset + k-chunk + sel*8; two contiguous 16B loads.
__device__ __forceinline__ v16bf load_frag_bf16(const __bf16* __restrict__ base) {
    v8bf lo = *(const v8bf*)(base);
    v8bf hi = *(const v8bf*)(base + 16);
    return __builtin_shufflevector(lo, hi, 0,1,2,3,4,5,6,7,8,9,10,11,12,13,14,15);
}

// Tensor Data Mover: DMA a 64-row x 256-col bf16 tile (row-major, stride 256)
// from global memory into LDS at byte offset lds_off.  D# per CDNA5 ISA ch.8.
__device__ __forceinline__ void tdm_load_tile_64x256_bf16(const __bf16* gptr, unsigned lds_off) {
    unsigned long long ga = (unsigned long long)(uintptr_t)gptr;
    v4u g0;
    g0[0] = 1u;                                              // count=1 (valid user D#)
    g0[1] = lds_off;                                         // lds_addr (bytes)
    g0[2] = (unsigned)(ga & 0xFFFFFFFFu);                    // global_addr[31:0]
    g0[3] = (unsigned)((ga >> 32) & 0x01FFFFFFu) | (2u << 30); // addr[56:32] | type=2
    v8i g1;
    g1[0] = 0x00010000;          // wg_mask=0 | data_size=1 (2 bytes)
    g1[1] = (int)(256u << 16);   // atomic_barrier_addr=0 | tensor_dim0[15:0]=256
    g1[2] = (int)(64u << 16);    // tensor_dim0[31:16]=0 | tensor_dim1[15:0]=64
    g1[3] = (int)(256u << 16);   // tensor_dim1[31:16]=0 | tile_dim0=256
    g1[4] = 64;                  // tile_dim1=64 | tile_dim2=0
    g1[5] = 256;                 // tensor_dim0_stride[31:0]=256
    g1[6] = 0;                   // stride0[47:32]=0 | tensor_dim1_stride lo=0
    g1[7] = 0;
    v4i g2 = {0, 0, 0, 0};       // 2-D tile: groups 2/3 unused
    v4i g3 = {0, 0, 0, 0};
#if __has_include(<hip/amd_detail/amd_gfx1250_TDM.h>)
    v8i gx = {0, 0, 0, 0, 0, 0, 0, 0};
    __builtin_amdgcn_tensor_load_to_lds(g0, g1, g2, g3, gx, 0);   // therock 6-arg form
#else
    __builtin_amdgcn_tensor_load_to_lds(g0, g1, g2, g3, 0);       // ROCm 7.2 5-arg form
#endif
}

// Monotonic float->uint mapping packed with index: u64 min == lexicographic
// (distance, index) min -> matches jnp.argmin tie-breaking (lowest index).
__device__ __forceinline__ unsigned long long pack_dist_idx(float v, int k) {
    unsigned int bits = __float_as_uint(v);
    bits = (bits & 0x80000000u) ? ~bits : (bits | 0x80000000u);
    return ((unsigned long long)bits << 32) | (unsigned int)k;
}

__device__ __forceinline__ unsigned long long shfl_xor_u64(unsigned long long v, int mask) {
    int lo = __shfl_xor((int)(v & 0xFFFFFFFFull), mask, 32);
    int hi = __shfl_xor((int)(v >> 32), mask, 32);
    return ((unsigned long long)(unsigned int)hi << 32) | (unsigned int)lo;
}

// ---------------- Kernel 1: codebook norms + argmin-slot init ----------------
__global__ void vq_init_kernel(const float* __restrict__ code,
                               float* __restrict__ c2,
                               unsigned long long* __restrict__ packed) {
    int i = blockIdx.x * blockDim.x + threadIdx.x;   // N_DIM threads total
    if (i < K_DIM) {
        const float* row = code + (size_t)i * R_DIM;
        float s = 0.f;
        for (int r = 0; r < R_DIM; ++r) s += row[r] * row[r];
        c2[i] = s;
    }
    if (i < N_DIM) packed[i] = ~0ull;
}

// ---------------- Kernel 2: encode  z = x @ W^T  (f32 WMMA, exact) ----------------
__global__ void vq_encode_kernel(const float* __restrict__ x,
                                 const float* __restrict__ W,
                                 float* __restrict__ z) {
    const int wave = (blockIdx.x * blockDim.x + threadIdx.x) >> 5;
    const int lane = threadIdx.x & 31;
    const int tilesN = R_DIM / 16;                    // 16
    const int m0 = (wave / tilesN) * 16;
    const int n0 = (wave % tilesN) * 16;
    const int half = lane >> 4;
    const int l16  = lane & 15;
    const int kSel = half * 2;

    const float* xrow = x + (size_t)(m0 + l16) * D_DIM + kSel;
    const float* wrow = W + (size_t)(n0 + l16) * D_DIM + kSel;

    v8f acc = {};
    for (int kk = 0; kk < D_DIM; kk += 4) {
        v2f a = *(const v2f*)(xrow + kk);
        v2f b = *(const v2f*)(wrow + kk);
        acc = wmma_f32x4(a, b, acc);
    }
    const int col = n0 + l16;
#pragma unroll
    for (int i = 0; i < 8; ++i) {
        int row = m0 + i + half * 8;
        z[(size_t)row * R_DIM + col] = acc[i];
    }
}

// ------- Kernel 3: bf16x3 split of z and code (hi + residual-lo) -------
__global__ void vq_split_kernel(const float* __restrict__ z,
                                const float* __restrict__ code,
                                __bf16* __restrict__ zh, __bf16* __restrict__ zl,
                                __bf16* __restrict__ ch, __bf16* __restrict__ cl) {
    size_t i = (size_t)blockIdx.x * blockDim.x + threadIdx.x;   // covers N*R
    if (i < (size_t)N_DIM * R_DIM) {
        float v = z[i];
        __bf16 h = (__bf16)v;
        zh[i] = h;
        zl[i] = (__bf16)(v - (float)h);
    }
    if (i < (size_t)K_DIM * R_DIM) {
        float v = code[i];
        __bf16 h = (__bf16)v;
        ch[i] = h;
        cl[i] = (__bf16)(v - (float)h);
    }
}

// ------- Kernel 4: distances + argmin (bf16x3 WMMA + TDM-staged code tile) -------
// Workgroup: 128 rows x 64 codes.  Code tile (hi+lo, 2x32KB) DMA'd to LDS once
// via TENSOR_LOAD_TO_LDS; 8 waves share it (8x less global B traffic).
// dist(n,k) ~ c2[k] - 2 * (z_n . c_k);  z.c ~= zh.ch + zh.cl + zl.ch (f32 acc).
__global__ void vq_dist_kernel(const __bf16* __restrict__ zh, const __bf16* __restrict__ zl,
                               const __bf16* __restrict__ ch, const __bf16* __restrict__ cl,
                               const float* __restrict__ c2,
                               unsigned long long* __restrict__ packed) {
    extern __shared__ __bf16 lds_tile[];              // [2][64][256] = 64 KB
    const int waveId = threadIdx.x >> 5;
    const int lane   = threadIdx.x & 31;
    const int tilesK = K_DIM / 64;                    // 128
    const int bm = blockIdx.x / tilesK;               // 0 .. N/128-1
    const int k0 = (blockIdx.x % tilesK) * 64;
    const int m0 = (bm * 8 + waveId) * 16;
    const int half = lane >> 4;
    const int l16  = lane & 15;
    const int sel8 = half * 8;

    // Wave 0 kicks off both DMA tiles, waits for TENSORcnt, then WG barrier.
    if (threadIdx.x < 32) {
        tdm_load_tile_64x256_bf16(ch + (size_t)k0 * R_DIM, 0u);
        tdm_load_tile_64x256_bf16(cl + (size_t)k0 * R_DIM, 64u * 256u * 2u);
        __builtin_amdgcn_s_wait_tensorcnt(0);
    }
    __syncthreads();

    const __bf16* lch = lds_tile;                     // local 64x256 hi tile
    const __bf16* lcl = lds_tile + 64 * 256;          // local 64x256 lo tile

    const __bf16* zhrow = zh + (size_t)(m0 + l16) * R_DIM + sel8;
    const __bf16* zlrow = zl + (size_t)(m0 + l16) * R_DIM + sel8;
    const __bf16* chrow = lch + (size_t)l16 * R_DIM + sel8;   // local code row l16
    const __bf16* clrow = lcl + (size_t)l16 * R_DIM + sel8;

    v8f acc0 = {}, acc1 = {}, acc2 = {}, acc3 = {};
    for (int kc = 0; kc < R_DIM; kc += 32) {
        v16bf ah = load_frag_bf16(zhrow + kc);
        v16bf al = load_frag_bf16(zlrow + kc);

        v16bf bh0 = load_frag_bf16(chrow + 0  * 16 * R_DIM + kc);
        v16bf bl0 = load_frag_bf16(clrow + 0  * 16 * R_DIM + kc);
        acc0 = wmma_bf16x32(ah, bh0, acc0);
        acc0 = wmma_bf16x32(ah, bl0, acc0);
        acc0 = wmma_bf16x32(al, bh0, acc0);

        v16bf bh1 = load_frag_bf16(chrow + 1 * 16 * R_DIM + kc);
        v16bf bl1 = load_frag_bf16(clrow + 1 * 16 * R_DIM + kc);
        acc1 = wmma_bf16x32(ah, bh1, acc1);
        acc1 = wmma_bf16x32(ah, bl1, acc1);
        acc1 = wmma_bf16x32(al, bh1, acc1);

        v16bf bh2 = load_frag_bf16(chrow + 2 * 16 * R_DIM + kc);
        v16bf bl2 = load_frag_bf16(clrow + 2 * 16 * R_DIM + kc);
        acc2 = wmma_bf16x32(ah, bh2, acc2);
        acc2 = wmma_bf16x32(ah, bl2, acc2);
        acc2 = wmma_bf16x32(al, bh2, acc2);

        v16bf bh3 = load_frag_bf16(chrow + 3 * 16 * R_DIM + kc);
        v16bf bl3 = load_frag_bf16(clrow + 3 * 16 * R_DIM + kc);
        acc3 = wmma_bf16x32(ah, bh3, acc3);
        acc3 = wmma_bf16x32(ah, bl3, acc3);
        acc3 = wmma_bf16x32(al, bh3, acc3);
    }

    const int kA = k0 + l16, kB = kA + 16, kC = kA + 32, kD = kA + 48;
    const float c2A = c2[kA], c2B = c2[kB], c2C = c2[kC], c2D = c2[kD];

#pragma unroll
    for (int i = 0; i < 8; ++i) {
        int row = m0 + i + half * 8;
        unsigned long long best = pack_dist_idx(c2A - 2.0f * acc0[i], kA);
        unsigned long long o;
        o = pack_dist_idx(c2B - 2.0f * acc1[i], kB); best = (o < best) ? o : best;
        o = pack_dist_idx(c2C - 2.0f * acc2[i], kC); best = (o < best) ? o : best;
        o = pack_dist_idx(c2D - 2.0f * acc3[i], kD); best = (o < best) ? o : best;
        for (int m = 8; m >= 1; m >>= 1) {
            o = shfl_xor_u64(best, m);
            best = (o < best) ? o : best;
        }
        if (l16 == 0) atomicMin(&packed[row], best);
    }
}

// ---- Kernel 5: gather q = code[idx], per-row (q-z)^2 partial, idx output ----
__global__ void vq_gather_kernel(const unsigned long long* __restrict__ packed,
                                 const float* __restrict__ code,
                                 const float* __restrict__ z,
                                 float* __restrict__ q,
                                 float* __restrict__ idx_out,
                                 float* __restrict__ cq_part) {
    __shared__ float sdata[R_DIM];
    const int n = blockIdx.x;
    const int t = threadIdx.x;                        // 256 == R_DIM
    const int idx = (int)(packed[n] & 0xFFFFFFFFull);
    const float qv = code[(size_t)idx * R_DIM + t];
    q[(size_t)n * R_DIM + t] = qv;
    const float diff = qv - z[(size_t)n * R_DIM + t];
    sdata[t] = diff * diff;
    __syncthreads();
    for (int s = R_DIM / 2; s > 0; s >>= 1) {
        if (t < s) sdata[t] += sdata[t + s];
        __syncthreads();
    }
    if (t == 0) { cq_part[n] = sdata[0]; idx_out[n] = (float)idx; }
}

// ---- Kernel 6: decode x_hat = q @ W (f32 WMMA), reduce (x_hat - x)^2 on the fly ----
__global__ void vq_decode_rec_kernel(const float* __restrict__ q,
                                     const float* __restrict__ W,
                                     const float* __restrict__ x,
                                     float* __restrict__ rec_part) {
    const int wave = (blockIdx.x * blockDim.x + threadIdx.x) >> 5;
    const int lane = threadIdx.x & 31;
    const int tilesD = D_DIM / 16;                    // 64
    const int m0 = (wave / tilesD) * 16;
    const int d0 = (wave % tilesD) * 16;
    const int half = lane >> 4;
    const int l16  = lane & 15;
    const int kSel = half * 2;

    const float* qrow = q + (size_t)(m0 + l16) * R_DIM + kSel;
    const int col = d0 + l16;

    v8f acc = {};
    for (int rr = 0; rr < R_DIM; rr += 4) {
        v2f a = *(const v2f*)(qrow + rr);
        v2f b;
        b.x = W[(size_t)(rr + kSel)     * D_DIM + col];
        b.y = W[(size_t)(rr + kSel + 1) * D_DIM + col];
        acc = wmma_f32x4(a, b, acc);
    }

    float s = 0.f;
#pragma unroll
    for (int i = 0; i < 8; ++i) {
        int row = m0 + i + half * 8;
        float diff = acc[i] - x[(size_t)row * D_DIM + col];
        s += diff * diff;
    }
    for (int m = 16; m >= 1; m >>= 1) s += __shfl_xor(s, m, 32);
    if (lane == 0) rec_part[wave] = s;
}

// ---------------- Kernel 7: deterministic final reduction ----------------
__global__ void vq_finalize_kernel(const float* __restrict__ rec_part, int nrec,
                                   const float* __restrict__ cq_part, int ncq,
                                   float* __restrict__ out) {
    __shared__ float sr[256], sc[256];
    const int t = threadIdx.x;
    float r = 0.f, c = 0.f;
    for (int i = t; i < nrec; i += 256) r += rec_part[i];
    for (int i = t; i < ncq;  i += 256) c += cq_part[i];
    sr[t] = r; sc[t] = c;
    __syncthreads();
    for (int s = 128; s > 0; s >>= 1) {
        if (t < s) { sr[t] += sr[t + s]; sc[t] += sc[t + s]; }
        __syncthreads();
    }
    if (t == 0) {
        float rec = sr[0] / ((float)N_DIM * (float)D_DIM);
        float cq  = sc[0] / ((float)N_DIM * (float)R_DIM);
        out[0] = rec + cq + BETA_F * cq;   // loss = rec + codebook + beta*commit
        out[1] = rec;
    }
}

extern "C" void kernel_launch(void* const* d_in, const int* in_sizes, int n_in,
                              void* d_out, int out_size, void* d_ws, size_t ws_size,
                              hipStream_t stream) {
    const float* x    = (const float*)d_in[0];   // [N, D]
    const float* W    = (const float*)d_in[1];   // [R, D]
    const float* code = (const float*)d_in[2];   // [K, R]

    float* out = (float*)d_out;
    // Output tuple layout: loss(1), rec(1), z(N*R), q(N*R), idx(N)
    float* z_out   = out + 2;
    float* q_out   = z_out + (size_t)N_DIM * R_DIM;
    float* idx_out = q_out + (size_t)N_DIM * R_DIM;

    // Workspace layout (16B-aligned offsets)
    char* ws = (char*)d_ws;
    float*              c2       = (float*)(ws);                    //  32 KB
    unsigned long long* packed   = (unsigned long long*)(ws + 32768);      // 256 KB
    float*              cq_part  = (float*)(ws + 294912);           // 128 KB
    float*              rec_part = (float*)(ws + 425984);           // 512 KB
    __bf16*             zh       = (__bf16*)(ws + 950272);          //  16 MB
    __bf16*             zl       = (__bf16*)(ws + 17727488);        //  16 MB
    __bf16*             ch       = (__bf16*)(ws + 34504704);        //   4 MB
    __bf16*             cl       = (__bf16*)(ws + 38699008);        //   4 MB

    const int wavesPerBlock = 8;     // 256 threads = 8 wave32s

    // 1) codebook norms + argmin init
    vq_init_kernel<<<N_DIM / 256, 256, 0, stream>>>(code, c2, packed);

    // 2) z = x @ W^T : (N/16)*(R/16) = 32768 tiles, one wave each
    vq_encode_kernel<<<(N_DIM / 16) * (R_DIM / 16) / wavesPerBlock, 256, 0, stream>>>(x, W, z_out);

    // 3) bf16x3 split of z and code
    vq_split_kernel<<<(N_DIM * (R_DIM / 256)), 256, 0, stream>>>(z_out, code, zh, zl, ch, cl);

    // 4) distances + argmin : (N/128)*(K/64) = 32768 workgroups, 64KB LDS each
    vq_dist_kernel<<<(N_DIM / 128) * (K_DIM / 64), 256, 65536, stream>>>(zh, zl, ch, cl, c2, packed);

    // 5) gather q, idx, commit partials : one block per row
    vq_gather_kernel<<<N_DIM, R_DIM, 0, stream>>>(packed, code, z_out, q_out, idx_out, cq_part);

    // 6) decode + rec partials : (N/16)*(D/16) = 131072 waves
    vq_decode_rec_kernel<<<(N_DIM / 16) * (D_DIM / 16) / wavesPerBlock, 256, 0, stream>>>(q_out, W, x, rec_part);

    // 7) deterministic scalar reduction
    vq_finalize_kernel<<<1, 256, 0, stream>>>(rec_part, (N_DIM / 16) * (D_DIM / 16),
                                              cq_part, N_DIM, out);
}
